// ExpertsChooseMaskedExpand_45913200394345
// MI455X (gfx1250) — matrix operand order repaired and checked by reading, not compiled, not run on hardware
//
#include <hip/hip_runtime.h>

// ---------------------------------------------------------------------------
// ExpertsChooseMaskedExpand for MI455X (gfx1250, wave32, WMMA, async-to-LDS)
//
//   Y[b,e,c,o]  = sum_i x[b,e,c,i] * W[e,o,i] + bias[o]          (GEMM1)
//   out[b,t,o]  = sum_{e,c} Y[b,e,c,o] * combine[b,t,e,c]        (GEMM2)
//
// bf16 WMMA (f32 accumulate) for both contractions. GEMM1 epilogue adds bias
// and transposes through LDS so the intermediate is K-contiguous per o-row:
// Yt[b][o][e*C+c] (bf16). Tiles are staged in LDS with
// global_load_async_to_lds_b128 + double buffering when the builtin exists.
// ---------------------------------------------------------------------------

typedef __attribute__((ext_vector_type(16))) __bf16 v16bf;
typedef __attribute__((ext_vector_type(4)))  __bf16 v4bf;
typedef __attribute__((ext_vector_type(8)))  float  v8f;

union FragU { uint4 u[2]; v16bf v; };

// problem constants
#define PB  4
#define PT  2048
#define PE  8
#define PC  512
#define PI  512
#define PO  4096
#define PEC (PE * PC)   // 4096

// tiling: block tile 128(M) x 256(N), 8 waves in 2x4 grid, wave tile 64x64
#define BM  128
#define BN  256
#define BK  32
#define LDT 40          // LDS row stride (bf16) for A/B tiles (pad 32 -> 40)
#define LDTT 136        // LDS row stride (bf16) for transpose tile
#define ASLAB (BM * LDT)            // 5120 bf16
#define BSLAB (BN * LDT)            // 10240 bf16
#define TILEBUF (ASLAB + BSLAB)     // 15360 bf16 per double-buffer slab

#ifndef __has_builtin
#define __has_builtin(x) 0
#endif
#if __has_builtin(__builtin_amdgcn_global_load_async_to_lds_b128) && \
    __has_builtin(__builtin_amdgcn_s_wait_asynccnt)
#define USE_ASYNC 1
#else
#define USE_ASYNC 0
#endif

#if USE_ASYNC
// per the compiler diagnostic, the builtin takes
//   (int __vector(4) addrspace(1)*, int __vector(4) addrspace(3)*, imm, imm)
typedef int v4i_vs __attribute__((vector_size(16)));
typedef __attribute__((address_space(1))) v4i_vs* as1_v4i;
typedef __attribute__((address_space(3))) v4i_vs* as3_v4i;

__device__ __forceinline__ void async_b128(const __bf16* g, __bf16* l) {
    __builtin_amdgcn_global_load_async_to_lds_b128((as1_v4i)g, (as3_v4i)l, 0, 0);
}
#endif

// ---------------------------------------------------------------------------
// f32 -> bf16 conversion (vectorized x4, grid-stride)
// ---------------------------------------------------------------------------
__global__ __launch_bounds__(256)
void cvt_f32_bf16_x4(const float* __restrict__ in, __bf16* __restrict__ out, int n4) {
    int i = blockIdx.x * blockDim.x + threadIdx.x;
    int stride = gridDim.x * blockDim.x;
    for (; i < n4; i += stride) {
        float4 f = ((const float4*)in)[i];
        v4bf o;
        o[0] = (__bf16)f.x; o[1] = (__bf16)f.y;
        o[2] = (__bf16)f.z; o[3] = (__bf16)f.w;
        *(v4bf*)(out + 4 * (size_t)i) = o;
    }
}

// ---------------------------------------------------------------------------
// shared GEMM machinery
// ---------------------------------------------------------------------------
__device__ __forceinline__ void compute_tile(const __bf16* As, const __bf16* Bs,
                                             int wm, int wn, int khalf, int ln,
                                             v8f acc[4][4])
{
    FragU a[4], b[4];
#pragma unroll
    for (int fm = 0; fm < 4; fm++) {
        const uint4* p = (const uint4*)&As[(wm * 64 + fm * 16 + ln) * LDT + khalf * 8];
        a[fm].u[0] = p[0];      // K = k0 .. k0+7      (k0 = 8*khalf)
        a[fm].u[1] = p[2];      // K = k0+16 .. k0+23
    }
#pragma unroll
    for (int fn = 0; fn < 4; fn++) {
        const uint4* p = (const uint4*)&Bs[(wn * 64 + fn * 16 + ln) * LDT + khalf * 8];
        b[fn].u[0] = p[0];      // K = k0 .. k0+7      (k0 = 8*khalf)
        b[fn].u[1] = p[2];      // K = k0+16 .. k0+23
    }
#pragma unroll
    for (int fm = 0; fm < 4; fm++)
#pragma unroll
        for (int fn = 0; fn < 4; fn++)
            acc[fm][fn] = __builtin_amdgcn_wmma_f32_16x16x32_bf16(
                false, a[fm].v, false, b[fn].v,
                (short)0, acc[fm][fn], false, false);
}

// A tile: BM rows x 32 bf16 (2 threads/row, 32B each);
// B tile: BN rows x 32 bf16 (1 thread/row, 64B)
__device__ __forceinline__ void gemm_mainloop(const __bf16* __restrict__ Ag,
                                              const __bf16* __restrict__ Bg,
                                              int ldk, int K,
                                              __bf16* smem, v8f acc[4][4])
{
    const int tid   = threadIdx.x;
    const int lane  = tid & 31;
    const int wave  = tid >> 5;
    const int wm    = wave >> 2;    // 0..1
    const int wn    = wave & 3;     // 0..3
    const int khalf = lane >> 4;
    const int ln    = lane & 15;

    const int arow = tid >> 1;          // 0..127
    const int aofs = (tid & 1) * 16;    // bf16 offset
    const int brow = tid;               // 0..255

#if USE_ASYNC
    const int nsteps = K / BK;
    // prologue: fill buffer 0
    {
        const __bf16* ga = Ag + (size_t)arow * ldk + aofs;
        __bf16* la = smem + arow * LDT + aofs;
        async_b128(ga, la); async_b128(ga + 8, la + 8);
        const __bf16* gb = Bg + (size_t)brow * ldk;
        __bf16* lb = smem + ASLAB + brow * LDT;
        async_b128(gb, lb);           async_b128(gb + 8,  lb + 8);
        async_b128(gb + 16, lb + 16); async_b128(gb + 24, lb + 24);
    }
    __builtin_amdgcn_s_wait_asynccnt(0);
    __syncthreads();
    for (int s = 0; s < nsteps; s++) {
        __bf16* cur = smem + (s & 1) * TILEBUF;
        if (s + 1 < nsteps) {   // prefetch next K-slab into other buffer
            __bf16* nxt = smem + ((s + 1) & 1) * TILEBUF;
            const int ks = (s + 1) * BK;
            const __bf16* ga = Ag + (size_t)arow * ldk + ks + aofs;
            __bf16* la = nxt + arow * LDT + aofs;
            async_b128(ga, la); async_b128(ga + 8, la + 8);
            const __bf16* gb = Bg + (size_t)brow * ldk + ks;
            __bf16* lb = nxt + ASLAB + brow * LDT;
            async_b128(gb, lb);           async_b128(gb + 8,  lb + 8);
            async_b128(gb + 16, lb + 16); async_b128(gb + 24, lb + 24);
        }
        compute_tile(cur, cur + ASLAB, wm, wn, khalf, ln, acc);
        __builtin_amdgcn_s_wait_asynccnt(0);
        __syncthreads();
    }
#else
    // synchronous fallback: single buffer, two barriers per K-step
    for (int ks = 0; ks < K; ks += BK) {
        __syncthreads();
        {
            const uint4* g = (const uint4*)(Ag + (size_t)arow * ldk + ks + aofs);
            uint4 d0 = g[0], d1 = g[1];
            *(uint4*)&smem[arow * LDT + aofs]     = d0;
            *(uint4*)&smem[arow * LDT + aofs + 8] = d1;
        }
        {
            const uint4* g = (const uint4*)(Bg + (size_t)brow * ldk + ks);
            uint4 d0 = g[0], d1 = g[1], d2 = g[2], d3 = g[3];
            __bf16* lb = smem + ASLAB + brow * LDT;
            *(uint4*)&lb[0]  = d0; *(uint4*)&lb[8]  = d1;
            *(uint4*)&lb[16] = d2; *(uint4*)&lb[24] = d3;
        }
        __syncthreads();
        compute_tile(smem, smem + ASLAB, wm, wn, khalf, ln, acc);
    }
#endif
}

// ---------------------------------------------------------------------------
// GEMM1: per (b,e):  D[c,o] = sum_i xb[.,c,i]*Wb[e,o,i] + bias[o]
// stores transposed:  Yt[b][o][e*PC + c]  (bf16)
// grid: 32 (b,e) * 4 (c-tiles) * 16 (o-tiles of 256) = 2048 blocks
// ---------------------------------------------------------------------------
__global__ __launch_bounds__(256)
void gemm1_kernel(const __bf16* __restrict__ Xb,   // [B,E,C,I]
                  const __bf16* __restrict__ Wb,   // [E,O,I]
                  const float*  __restrict__ bias, // [O]
                  __bf16*       __restrict__ Yt)   // [B,O,E*C]
{
    // max(double-buffered A/B tiles = 30720, transpose tile 256*136 = 34816)
    __shared__ __align__(16) __bf16 smem[BN * LDTT];

    const int bx  = blockIdx.x;
    const int be  = bx >> 6;        // 0..31
    const int rem = bx & 63;
    const int tm  = rem & 3;        // c-tile: C/128 = 4
    const int tn  = rem >> 2;       // o-tile: O/256 = 16
    const int b   = be >> 3;
    const int e   = be & 7;

    const int tid   = threadIdx.x;
    const int lane  = tid & 31;
    const int wave  = tid >> 5;
    const int wm    = wave >> 2;
    const int wn    = wave & 3;
    const int khalf = lane >> 4;
    const int ln    = lane & 15;

    const __bf16* Ag = Xb + ((size_t)(b * PE + e) * PC + (size_t)tm * BM) * PI;
    const __bf16* Bg = Wb + ((size_t)e * PO + (size_t)tn * BN) * PI;

    v8f acc[4][4];
#pragma unroll
    for (int i = 0; i < 4; i++)
#pragma unroll
        for (int j = 0; j < 4; j++)
            acc[i][j] = (v8f){0.f,0.f,0.f,0.f,0.f,0.f,0.f,0.f};

    gemm_mainloop(Ag, Bg, PI, PI, smem, acc);

    // ---- epilogue: bias add, bf16 convert, transpose via LDS, store Yt ----
    __syncthreads();
    __bf16* Ts = smem; // [256 (o)][LDTT (c)]
#pragma unroll
    for (int fm = 0; fm < 4; fm++) {
#pragma unroll
        for (int fn = 0; fn < 4; fn++) {
            const int n_l = wn * 64 + fn * 16 + ln;
            const float bv = bias[tn * BN + n_l];
#pragma unroll
            for (int r2 = 0; r2 < 8; r2++) {
                const int m_l = wm * 64 + fm * 16 + khalf * 8 + r2;
                Ts[n_l * LDTT + m_l] = (__bf16)(acc[fm][fn][r2] + bv);
            }
        }
    }
    __syncthreads();
    {
        const int orow = tid;                // o row in tile (0..255)
        const size_t dst = ((size_t)b * PO + (size_t)tn * BN + orow) * PEC
                         + (size_t)e * PC + (size_t)tm * BM;
        const uint4* s = (const uint4*)&Ts[orow * LDTT];
        uint4* d = (uint4*)&Yt[dst];
#pragma unroll
        for (int q = 0; q < 16; q++) d[q] = s[q];   // 128 bf16 per row
    }
}

// ---------------------------------------------------------------------------
// GEMM2: per b:  out[t,o] = sum_k combine_b[t,k] * Yt_b[o,k],  k = e*C+c
// grid: 4 (b) * 16 (t-tiles) * 16 (o-tiles of 256) = 1024 blocks
// ---------------------------------------------------------------------------
__global__ __launch_bounds__(256)
void gemm2_kernel(const __bf16* __restrict__ Ab,  // [B,T,EC] bf16 combine
                  const __bf16* __restrict__ Yt,  // [B,O,EC]
                  float*        __restrict__ Out) // [B,T,O]
{
    __shared__ __align__(16) __bf16 smem[2 * TILEBUF];

    const int bx  = blockIdx.x;
    const int b   = bx >> 8;       // /256
    const int rem = bx & 255;
    const int tm  = rem & 15;      // t-tile: T/128 = 16
    const int tn  = rem >> 4;      // o-tile: O/256 = 16

    const int tid   = threadIdx.x;
    const int lane  = tid & 31;
    const int wave  = tid >> 5;
    const int wm    = wave >> 2;
    const int wn    = wave & 3;
    const int khalf = lane >> 4;
    const int ln    = lane & 15;

    const __bf16* Ag = Ab + ((size_t)b * PT + (size_t)tm * BM) * PEC;
    const __bf16* Bg = Yt + ((size_t)b * PO + (size_t)tn * BN) * PEC;

    v8f acc[4][4];
#pragma unroll
    for (int i = 0; i < 4; i++)
#pragma unroll
        for (int j = 0; j < 4; j++)
            acc[i][j] = (v8f){0.f,0.f,0.f,0.f,0.f,0.f,0.f,0.f};

    gemm_mainloop(Ag, Bg, PEC, PEC, smem, acc);

    // ---- epilogue: direct coalesced f32 stores ----
#pragma unroll
    for (int fm = 0; fm < 4; fm++) {
#pragma unroll
        for (int fn = 0; fn < 4; fn++) {
            const int n = tn * BN + wn * 64 + fn * 16 + ln;
#pragma unroll
            for (int r2 = 0; r2 < 8; r2++) {
                const int m = tm * BM + wm * 64 + fm * 16 + khalf * 8 + r2;
                Out[((size_t)b * PT + m) * PO + n] = acc[fm][fn][r2];
            }
        }
    }
}

// ---------------------------------------------------------------------------
// launcher
// inputs: d_in[0]=x [B,E,C,I] f32, d_in[1]=combine [B,T,E,C] f32,
//         d_in[2]=W [E,O,I] f32, d_in[3]=bias [O] f32
// output: d_out [B,T,O] f32
// ---------------------------------------------------------------------------
extern "C" void kernel_launch(void* const* d_in, const int* in_sizes, int n_in,
                              void* d_out, int out_size, void* d_ws, size_t ws_size,
                              hipStream_t stream) {
    const float* x    = (const float*)d_in[0];
    const float* comb = (const float*)d_in[1];
    const float* W    = (const float*)d_in[2];
    const float* bias = (const float*)d_in[3];
    float* out = (float*)d_out;

    char* ws = (char*)d_ws;
    __bf16* xb = (__bf16*)(ws);                              // 16 MB
    __bf16* Wb = (__bf16*)(ws + (size_t)16  * 1024 * 1024);  // 32 MB
    __bf16* Ab = (__bf16*)(ws + (size_t)48  * 1024 * 1024);  // 64 MB
    __bf16* Yt = (__bf16*)(ws + (size_t)112 * 1024 * 1024);  // 128 MB

    const int n4_x = (PB * PE * PC * PI) / 4;       // 2,097,152
    const int n4_W = (PE * PO * PI) / 4;            // 4,194,304
    const int n4_A = (PB * PT * PE * PC) / 4;       // 8,388,608

    cvt_f32_bf16_x4<<<2048, 256, 0, stream>>>(x,    xb, n4_x);
    cvt_f32_bf16_x4<<<4096, 256, 0, stream>>>(W,    Wb, n4_W);
    cvt_f32_bf16_x4<<<8192, 256, 0, stream>>>(comb, Ab, n4_A);

    gemm1_kernel<<<2048, 256, 0, stream>>>(xb, Wb, bias, Yt);
    gemm2_kernel<<<1024, 256, 0, stream>>>(Ab, Yt, out);
}
